// SelfAttention_59124519797200
// MI455X (gfx1250) — compile-verified, matrix operand order checked by reading, not run on hardware
//
#include <hip/hip_runtime.h>
#include <hip/hip_bf16.h>
#include <math.h>

// Problem constants (match reference)
#define NTOK   2048
#define DMODEL 1024
#define NHEAD  16
#define HDIM   64
#define NEG_INF (-3.0e38f)
// 1/sqrt(HD) * log2(e): softmax done in log2 domain (v_exp_f32 is native exp2)
#define SCL_LOG2E 0.1803368801111204f

#define BK      64
#define LDSPAD  72   // row stride (elems): 144B rows -> conflict-minimal b128 frag loads

#define AS1 __attribute__((address_space(1)))
#define AS3 __attribute__((address_space(3)))

// Matches the builtin's parameter type: vector of 4 ints (16 bytes).
typedef __attribute__((__vector_size__(4 * sizeof(int)))) int i32x4v;

#if defined(__gfx1250__) && __has_builtin(__builtin_amdgcn_global_load_async_to_lds_b128)
#define HAS_ASYNC_LDS 1
#else
#define HAS_ASYNC_LDS 0
#endif

__device__ __forceinline__ void wait_asynccnt0() {
#if HAS_ASYNC_LDS
#if __has_builtin(__builtin_amdgcn_s_wait_asynccnt)
    __builtin_amdgcn_s_wait_asynccnt(0);
#else
    asm volatile("s_wait_asynccnt 0x0" ::: "memory");
#endif
#endif
}

// Copy one 16-byte chunk global -> LDS. Uses the CDNA5 async-to-LDS DMA path
// (ASYNCcnt) when available; falls back to a VGPR round-trip copy otherwise.
__device__ __forceinline__ void copy16_g2l(const __bf16* gsrc, __bf16* ldst) {
#if HAS_ASYNC_LDS
    __builtin_amdgcn_global_load_async_to_lds_b128(
        (AS1 i32x4v*)gsrc, (AS3 i32x4v*)ldst, 0, 0);
#else
    *reinterpret_cast<uint4*>(ldst) = *reinterpret_cast<const uint4*>(gsrc);
#endif
}

typedef __attribute__((ext_vector_type(16))) __bf16 bf16x16;
typedef __attribute__((ext_vector_type(8)))  float  f32x8;

union FragAB {
    bf16x16 v;
    uint4   q[2];
};

// Load a 16x32 bf16 WMMA A/B fragment from a row-major [rows][LDSPAD] bf16 LDS
// tile. CDNA5 16-bit operand layout: lanes 0-15 hold K {0..7,16..23}, lanes
// 16-31 hold K {8..15,24..31}; each half is two contiguous 16-byte chunks.
__device__ __forceinline__ FragAB load_frag(const __bf16* base, int row0, int k0,
                                            int lane) {
    FragAB f;
    const int r  = row0 + (lane & 15);
    const int kb = k0 + ((lane & 16) ? 8 : 0);
    const __bf16* p = base + r * LDSPAD + kb;
    f.q[0] = *reinterpret_cast<const uint4*>(p);
    f.q[1] = *reinterpret_cast<const uint4*>(p + 16);
    return f;
}

__device__ __forceinline__ f32x8 f32x8_zero() {
    f32x8 z = {0.f, 0.f, 0.f, 0.f, 0.f, 0.f, 0.f, 0.f};
    return z;
}

// ---------------------------------------------------------------------------
// fp32 -> bf16 elementwise (vectorized, n % 8 == 0)
// ---------------------------------------------------------------------------
__global__ void f32_to_bf16_kernel(const float* __restrict__ src,
                                   __bf16* __restrict__ dst, int n) {
    int i = (blockIdx.x * blockDim.x + threadIdx.x) * 8;
    if (i < n) {
        float4 a = *reinterpret_cast<const float4*>(src + i);
        float4 b = *reinterpret_cast<const float4*>(src + i + 4);
        union { uint4 u; __bf16 e[8]; } cv;
        cv.e[0] = (__bf16)a.x; cv.e[1] = (__bf16)a.y;
        cv.e[2] = (__bf16)a.z; cv.e[3] = (__bf16)a.w;
        cv.e[4] = (__bf16)b.x; cv.e[5] = (__bf16)b.y;
        cv.e[6] = (__bf16)b.z; cv.e[7] = (__bf16)b.w;
        *reinterpret_cast<uint4*>(dst + i) = cv.u;
    }
}

// ---------------------------------------------------------------------------
// fp32 [rows][cols] -> bf16 transposed [cols][rows], tiled through LDS.
// Grid (cols/32, rows/32), block (32, 8).
// ---------------------------------------------------------------------------
__global__ void transpose_f32_to_bf16_kernel(const float* __restrict__ src,
                                             __bf16* __restrict__ dst,
                                             int rows, int cols) {
    __shared__ float tile[32][33];
    const int bx = blockIdx.x * 32;   // col base in src
    const int by = blockIdx.y * 32;   // row base in src
    const int tx = threadIdx.x, ty = threadIdx.y;
#pragma unroll
    for (int j = 0; j < 32; j += 8)
        tile[ty + j][tx] = src[(size_t)(by + ty + j) * cols + bx + tx];
    __syncthreads();
#pragma unroll
    for (int j = 0; j < 32; j += 8)
        dst[(size_t)(bx + ty + j) * rows + by + tx] = (__bf16)tile[tx][ty + j];
}

// ---------------------------------------------------------------------------
// C[M,N] = A[M,K]bf16 @ W[K,N] + bias[N], where Bt = W^T is passed as
// [N][K] bf16 so the B-tile load is a straight vectorized copy.
// 128x128 tile / WG, BK=64, register double-buffered global loads.
// 8 waves: (wm 0..1) x (wn 0..3); each wave: 64x32 via 4x2 accumulators.
// OUTMODE: 0 = f32 out [M][N], 1 = bf16 out [M][N], 2 = bf16 out transposed [N][M]
// ---------------------------------------------------------------------------
template <int OUTMODE>
__global__ __launch_bounds__(256)
void gemm_bf16_kernel(const __bf16* __restrict__ A, const __bf16* __restrict__ Bt,
                      const float* __restrict__ bias, void* __restrict__ Cout,
                      int M, int Nn, int Kk) {
    __shared__ __align__(16) __bf16 As[128][LDSPAD];
    __shared__ __align__(16) __bf16 Bs[128][LDSPAD];

    const int tid  = threadIdx.x;
    const int lane = tid & 31;
    const int wave = tid >> 5;
    const int wm   = wave >> 2;   // 0..1
    const int wn   = wave & 3;    // 0..3
    const int m0   = blockIdx.y * 128;
    const int n0   = blockIdx.x * 128;

    f32x8 acc[4][2];
#pragma unroll
    for (int i = 0; i < 4; ++i)
#pragma unroll
        for (int j = 0; j < 2; ++j) acc[i][j] = f32x8_zero();

    // 128x64 tile = 1024 chunks of 8 bf16; 4 chunks/thread.
    uint4 pa[4], pb[4];
#pragma unroll
    for (int i = 0; i < 4; ++i) {
        const int idx = tid + i * 256;
        const int r = idx >> 3, c = (idx & 7) * 8;
        pa[i] = *reinterpret_cast<const uint4*>(A  + (size_t)(m0 + r) * Kk + c);
        pb[i] = *reinterpret_cast<const uint4*>(Bt + (size_t)(n0 + r) * Kk + c);
    }

    for (int k0 = 0; k0 < Kk; k0 += BK) {
#pragma unroll
        for (int i = 0; i < 4; ++i) {
            const int idx = tid + i * 256;
            const int r = idx >> 3, c = (idx & 7) * 8;
            *reinterpret_cast<uint4*>(&As[r][c]) = pa[i];
            *reinterpret_cast<uint4*>(&Bs[r][c]) = pb[i];
        }
        __syncthreads();

        // Issue next tile's global loads so HBM latency overlaps the WMMAs.
        if (k0 + BK < Kk) {
#pragma unroll
            for (int i = 0; i < 4; ++i) {
                const int idx = tid + i * 256;
                const int r = idx >> 3, c = (idx & 7) * 8;
                pa[i] = *reinterpret_cast<const uint4*>(A  + (size_t)(m0 + r) * Kk + k0 + BK + c);
                pb[i] = *reinterpret_cast<const uint4*>(Bt + (size_t)(n0 + r) * Kk + k0 + BK + c);
            }
        }

#pragma unroll
        for (int ks = 0; ks < 2; ++ks) {
            FragAB af[4], bf[2];
#pragma unroll
            for (int mt = 0; mt < 4; ++mt)
                af[mt] = load_frag(&As[0][0], wm * 64 + mt * 16, ks * 32, lane);
#pragma unroll
            for (int nt = 0; nt < 2; ++nt)
                bf[nt] = load_frag(&Bs[0][0], wn * 32 + nt * 16, ks * 32, lane);
#pragma unroll
            for (int mt = 0; mt < 4; ++mt)
#pragma unroll
                for (int nt = 0; nt < 2; ++nt)
                    acc[mt][nt] = __builtin_amdgcn_wmma_f32_16x16x32_bf16(
                        false, af[mt].v, false, bf[nt].v, (short)0, acc[mt][nt],
                        false, false);
        }
        __syncthreads();
    }

    // Epilogue: C layout — VGPR r is row r (lanes 0-15) / r+8 (lanes 16-31),
    // col = (lane & 15) within the 16-wide n-tile.
    const int rbase = (lane & 16) ? 8 : 0;
    const int ncol  = lane & 15;
#pragma unroll
    for (int mt = 0; mt < 4; ++mt) {
#pragma unroll
        for (int nt = 0; nt < 2; ++nt) {
            const int col = n0 + wn * 32 + nt * 16 + ncol;
            const float bv = bias[col];
#pragma unroll
            for (int r = 0; r < 8; ++r) {
                const int row = m0 + wm * 64 + mt * 16 + rbase + r;
                const float v = acc[mt][nt][r] + bv;
                if (OUTMODE == 0)
                    ((float*)Cout)[(size_t)row * Nn + col] = v;
                else if (OUTMODE == 1)
                    ((__bf16*)Cout)[(size_t)row * Nn + col] = (__bf16)v;
                else
                    ((__bf16*)Cout)[(size_t)col * M + row] = (__bf16)v;
            }
        }
    }
}

// ---------------------------------------------------------------------------
// Flash-attention (causal). Grid: (N/64 q-blocks, H heads). Block: 128 thr
// (4 waves); each wave owns 16 query rows, streams 64-key blocks with online
// softmax in the log2 domain. Diagonal block is peeled so the main loop is
// mask-free. Softmax denominator accumulated with a P @ ones WMMA. Q/K/V
// tiles staged with async global->LDS DMA (ASYNCcnt) when available.
// Vtg is V^T ([D][N] bf16).
// ---------------------------------------------------------------------------
__global__ __launch_bounds__(128)
void attn_kernel(const __bf16* __restrict__ Q, const __bf16* __restrict__ Kmat,
                 const __bf16* __restrict__ Vtg, __bf16* __restrict__ O) {
    __shared__ __align__(16) __bf16 Qs[64][LDSPAD];       // [q][hd]
    __shared__ __align__(16) __bf16 Ks[64][LDSPAD];       // [key][hd]
    __shared__ __align__(16) __bf16 Vts[64][LDSPAD];      // [hd][key]
    __shared__ __align__(16) __bf16 Ps[4][16][LDSPAD];    // per-wave P [q][key]

    const int tid  = threadIdx.x;
    const int lane = tid & 31;
    const int wave = tid >> 5;
    const int iq = blockIdx.x;      // query block index (0..31)
    const int h  = blockIdx.y;      // head
    const int q0 = iq * 64;
    const int hc = h * HDIM;

    // Load Q tile (64x64 bf16): 512 chunks of 8 bf16, 4 per thread
    for (int i = tid; i < 64 * 8; i += 128) {
        const int r = i >> 3, c = (i & 7) * 8;
        copy16_g2l(Q + (size_t)(q0 + r) * DMODEL + hc + c, &Qs[r][c]);
    }
    wait_asynccnt0();
    __syncthreads();

    // Q fragments are invariant across key blocks: load once.
    FragAB qa[2];
    qa[0] = load_frag(&Qs[0][0], wave * 16, 0, lane);
    qa[1] = load_frag(&Qs[0][0], wave * 16, 32, lane);

    // All-ones B fragment: P @ ones gives per-row sums in C layout.
    FragAB ones;
#pragma unroll
    for (int i = 0; i < 16; ++i) ones.v[i] = (__bf16)1.0f;

    const int rbase = (lane & 16) ? 8 : 0;
    const int ncol  = lane & 15;

    float m_i[8];
    f32x8 accO[4];
    f32x8 lacc;               // running softmax denominator (row sums)
#pragma unroll
    for (int r = 0; r < 8; ++r) m_i[r] = NEG_INF;
#pragma unroll
    for (int t = 0; t < 4; ++t) accO[t] = f32x8_zero();
    lacc = f32x8_zero();

    auto process_block = [&](int kb, bool diag) {
        const int k0r = kb * 64;
        __syncthreads();   // all waves done reading Ks/Vts/Ps from last round
        for (int i = tid; i < 64 * 8; i += 128) {
            const int r = i >> 3, c = (i & 7) * 8;
            copy16_g2l(Kmat + (size_t)(k0r + r) * DMODEL + hc + c, &Ks[r][c]);
            copy16_g2l(Vtg  + (size_t)(hc + r) * NTOK  + k0r + c, &Vts[r][c]);
        }
        wait_asynccnt0();
        __syncthreads();

        // Prefetch next key block while we compute (global_prefetch_b8).
        if (kb < iq) {
            const int nr = k0r + 64;
            if (tid < 64)
                __builtin_prefetch(Kmat + (size_t)(nr + tid) * DMODEL + hc, 0, 3);
            else
                __builtin_prefetch(Vtg + (size_t)(hc + tid - 64) * NTOK + nr, 0, 3);
        }

        // S = Q @ K^T  (16 q-rows x 64 keys per wave, f32 accum)
        f32x8 s[4];
#pragma unroll
        for (int t = 0; t < 4; ++t) {
            f32x8 sa = f32x8_zero();
#pragma unroll
            for (int ks = 0; ks < 2; ++ks) {
                FragAB bfr = load_frag(&Ks[0][0], t * 16, ks * 32, lane);
                sa = __builtin_amdgcn_wmma_f32_16x16x32_bf16(
                    false, qa[ks].v, false, bfr.v, (short)0, sa, false, false);
            }
            s[t] = sa;
        }

        // Scale into log2 domain; causal mask only on the (peeled) diagonal.
        float rowmax[8];
#pragma unroll
        for (int r = 0; r < 8; ++r) {
            float rm = NEG_INF;
#pragma unroll
            for (int t = 0; t < 4; ++t) {
                float v = s[t][r] * SCL_LOG2E;
                if (diag) {
                    const int key = (t << 4) + ncol;
                    const int qr  = (wave << 4) + rbase + r;
                    if (key > qr) v = NEG_INF;
                }
                s[t][r] = v;
                rm = fmaxf(rm, v);
            }
            rowmax[r] = rm;
        }
        // Cross-lane row max: level-major so the 8 bpermutes per level batch.
#pragma unroll
        for (int off = 1; off < 16; off <<= 1)
#pragma unroll
            for (int r = 0; r < 8; ++r)
                rowmax[r] = fmaxf(rowmax[r], __shfl_xor(rowmax[r], off, 32));

        float alpha[8];
#pragma unroll
        for (int r = 0; r < 8; ++r) {
            const float mnew = fmaxf(m_i[r], rowmax[r]);
            alpha[r] = exp2f(m_i[r] - mnew);
            m_i[r]   = mnew;
#pragma unroll
            for (int t = 0; t < 4; ++t)
                s[t][r] = exp2f(s[t][r] - mnew);
        }
        // Rescale running O and denominator by alpha.
#pragma unroll
        for (int t = 0; t < 4; ++t)
#pragma unroll
            for (int r = 0; r < 8; ++r) accO[t][r] *= alpha[r];
#pragma unroll
        for (int r = 0; r < 8; ++r) lacc[r] *= alpha[r];

        // Stage P (bf16) into wave-private LDS to re-enter WMMA A-layout.
#pragma unroll
        for (int t = 0; t < 4; ++t)
#pragma unroll
            for (int r = 0; r < 8; ++r)
                Ps[wave][rbase + r][(t << 4) + ncol] = (__bf16)s[t][r];
        __syncthreads();

        FragAB pa[2];
        pa[0] = load_frag(&Ps[wave][0][0], 0, 0, lane);
        pa[1] = load_frag(&Ps[wave][0][0], 0, 32, lane);

        // Denominator: lacc += P @ ones (row sums, no shuffles needed).
        lacc = __builtin_amdgcn_wmma_f32_16x16x32_bf16(
            false, pa[0].v, false, ones.v, (short)0, lacc, false, false);
        lacc = __builtin_amdgcn_wmma_f32_16x16x32_bf16(
            false, pa[1].v, false, ones.v, (short)0, lacc, false, false);

        // O += P @ V
#pragma unroll
        for (int t = 0; t < 4; ++t) {
#pragma unroll
            for (int ks = 0; ks < 2; ++ks) {
                FragAB bfr = load_frag(&Vts[0][0], t * 16, ks * 32, lane);
                accO[t] = __builtin_amdgcn_wmma_f32_16x16x32_bf16(
                    false, pa[ks].v, false, bfr.v, (short)0, accO[t], false, false);
            }
        }
    };

    for (int kb = 0; kb < iq; ++kb) process_block(kb, false);  // mask-free body
    process_block(iq, true);                                   // peeled diagonal

    // Normalize (one reciprocal per row) and write O (bf16 feed for out-proj).
    float inv[8];
#pragma unroll
    for (int r = 0; r < 8; ++r) inv[r] = 1.0f / lacc[r];
#pragma unroll
    for (int t = 0; t < 4; ++t) {
#pragma unroll
        for (int r = 0; r < 8; ++r) {
            const float v = accO[t][r] * inv[r];
            const int row = q0 + wave * 16 + rbase + r;
            const int col = hc + (t << 4) + ncol;
            O[(size_t)row * DMODEL + col] = (__bf16)v;
        }
    }
}

// ---------------------------------------------------------------------------
// Host launch
// ---------------------------------------------------------------------------
extern "C" void kernel_launch(void* const* d_in, const int* in_sizes, int n_in,
                              void* d_out, int out_size, void* d_ws, size_t ws_size,
                              hipStream_t stream) {
    const float* x  = (const float*)d_in[0];
    const float* Wq = (const float*)d_in[1];
    const float* bq = (const float*)d_in[2];
    const float* Wk = (const float*)d_in[3];
    const float* bk = (const float*)d_in[4];
    const float* Wv = (const float*)d_in[5];
    const float* bv = (const float*)d_in[6];
    const float* Wo = (const float*)d_in[7];
    const float* bo = (const float*)d_in[8];

    char* ws = (char*)d_ws;
    const size_t MB = 1024 * 1024;
    __bf16* xbf  = (__bf16*)(ws + 0);        // x bf16 [N][D]      4 MB
    __bf16* wqt  = (__bf16*)(ws + 4  * MB);  // Wq^T bf16 [D][D]   2 MB
    __bf16* wkt  = (__bf16*)(ws + 6  * MB);
    __bf16* wvt  = (__bf16*)(ws + 8  * MB);
    __bf16* wot  = (__bf16*)(ws + 10 * MB);
    __bf16* qb   = (__bf16*)(ws + 12 * MB);  // Q  [N][D]          4 MB
    __bf16* kb   = (__bf16*)(ws + 16 * MB);  // K  [N][D]
    __bf16* vtb  = (__bf16*)(ws + 20 * MB);  // V^T [D][N]
    __bf16* ab   = (__bf16*)(ws + 24 * MB);  // attn out [N][D]

    const int NE = NTOK * DMODEL;    // 2M
    f32_to_bf16_kernel<<<(NE / 8 + 255) / 256, 256, 0, stream>>>(x, xbf, NE);

    dim3 tb(32, 8), tg(DMODEL / 32, DMODEL / 32);
    transpose_f32_to_bf16_kernel<<<tg, tb, 0, stream>>>(Wq, wqt, DMODEL, DMODEL);
    transpose_f32_to_bf16_kernel<<<tg, tb, 0, stream>>>(Wk, wkt, DMODEL, DMODEL);
    transpose_f32_to_bf16_kernel<<<tg, tb, 0, stream>>>(Wv, wvt, DMODEL, DMODEL);
    transpose_f32_to_bf16_kernel<<<tg, tb, 0, stream>>>(Wo, wot, DMODEL, DMODEL);

    dim3 gg(DMODEL / 128, NTOK / 128);  // (8, 16)
    gemm_bf16_kernel<1><<<gg, 256, 0, stream>>>(xbf, wqt, bq, qb,  NTOK, DMODEL, DMODEL);
    gemm_bf16_kernel<1><<<gg, 256, 0, stream>>>(xbf, wkt, bk, kb,  NTOK, DMODEL, DMODEL);
    gemm_bf16_kernel<2><<<gg, 256, 0, stream>>>(xbf, wvt, bv, vtb, NTOK, DMODEL, DMODEL);

    attn_kernel<<<dim3(NTOK / 64, NHEAD), 128, 0, stream>>>(qb, kb, vtb, ab);

    gemm_bf16_kernel<0><<<gg, 256, 0, stream>>>(ab, wot, bo, d_out, NTOK, DMODEL, DMODEL);
}
